// Self_Attn_37769942401361
// MI455X (gfx1250) — matrix operand order repaired and checked by reading, not compiled, not run on hardware
//
#include <hip/hip_runtime.h>
#include <hip/hip_bf16.h>

// ---------------------------------------------------------------------------
// Self-attention (B=16, N=4096, C=512, D=64) for MI455X / gfx1250 (wave32).
// bf16 WMMA (v_wmma_f32_16x16x32_bf16) for all GEMMs, fp32 softmax stats,
// two-pass (recompute) softmax writing the 1.07GB attention matrix straight
// to d_out. Flat LDS tile staging uses CDNA5 async global->LDS DMA
// (global_load_async_to_lds_b128 + s_wait_asynccnt).
// ---------------------------------------------------------------------------

#define BB 16
#define NN 4096
#define CC 512
#define DD 64

typedef __attribute__((ext_vector_type(16))) __bf16 bf16x16;
typedef __attribute__((ext_vector_type(8)))  float  v8f;

static __device__ __forceinline__ unsigned short f32_to_bf16_bits(float f) {
  unsigned int u = __builtin_bit_cast(unsigned int, f);
  u += 0x7FFFu + ((u >> 16) & 1u);          // round-to-nearest-even
  return (unsigned short)(u >> 16);
}

// Async DMA: 16 bytes global -> LDS, tracked by ASYNCcnt (ISA 15.18.3 op 98).
static __device__ __forceinline__ void async_load_b128(void* lds, const void* g) {
  unsigned ldsOff = (unsigned)(unsigned long long)lds;   // addr[31:0] = LDS offset
  asm volatile("global_load_async_to_lds_b128 %0, %1, off"
               :: "v"(ldsOff), "v"((unsigned long long)g)
               : "memory");
}
static __device__ __forceinline__ void wait_async0() {
  asm volatile("s_wait_asynccnt 0" ::: "memory");
}

// 16 contiguous bf16 (B-fragment: K = 16h + 0..15 at column lane)
static __device__ __forceinline__ bf16x16 frag_contig16(const unsigned short* p) {
  union { unsigned int u[8]; bf16x16 v; } r;
  const unsigned int* q = (const unsigned int*)p;
#pragma unroll
  for (int i = 0; i < 8; ++i) r.u[i] = q[i];
  return r.v;
}

// A-fragment: two 8-element K chunks (callers pass K = 16h+0..7, 16h+8..15: contiguous, same as B)
static __device__ __forceinline__ bf16x16 frag_a8_8(const unsigned short* p0,
                                                    const unsigned short* p1) {
  union { unsigned int u[8]; bf16x16 v; } r;
  const unsigned int* q0 = (const unsigned int*)p0;
  const unsigned int* q1 = (const unsigned int*)p1;
#pragma unroll
  for (int i = 0; i < 4; ++i) r.u[i] = q0[i];
#pragma unroll
  for (int i = 0; i < 4; ++i) r.u[4 + i] = q1[i];
  return r.v;
}

static __device__ __forceinline__ v8f wmma_bf16(bf16x16 a, bf16x16 b, v8f c) {
  return __builtin_amdgcn_wmma_f32_16x16x32_bf16(false, a, false, b, (short)0, c,
                                                 false, false);
}

// ---------------------------------------------------------------------------
// Kernel 0: fp32 -> bf16 cast (grid-stride)
// ---------------------------------------------------------------------------
__global__ void cast_f32_bf16(const float* __restrict__ src,
                              unsigned short* __restrict__ dst, long n) {
  long i = (long)blockIdx.x * blockDim.x + threadIdx.x;
  long stride = (long)gridDim.x * blockDim.x;
  for (; i < n; i += stride) dst[i] = f32_to_bf16_bits(src[i]);
}

// ---------------------------------------------------------------------------
// Kernel 1: C = A[M,K](bf16) * W[K,Nc](bf16) + bias, bf16 out.
// Tile 128x64, 8 waves (16 rows each), Ktile=32. A tile staged via async DMA.
// ---------------------------------------------------------------------------
__global__ __launch_bounds__(256) void gemm_bias_bf16(
    const unsigned short* __restrict__ A, const unsigned short* __restrict__ W,
    const float* __restrict__ bias, unsigned short* __restrict__ Out,
    int M, int K, int Nc) {
  __shared__ unsigned short As[128 * 32];   // [row][k]
  __shared__ unsigned short Ws[64 * 32];    // transposed: [n][k]
  const int tid = threadIdx.x, lane = tid & 31, w = tid >> 5;
  const int r = lane & 15, h = lane >> 4;
  const int rowBase = blockIdx.x * 128;
  const int colBase = blockIdx.y * 64;
  v8f acc[4] = {};

  for (int kt = 0; kt < K; kt += 32) {
#pragma unroll
    for (int j = 0; j < 2; ++j) {             // 512 x 16B chunks = 128x32 bf16
      int idx = tid + j * 256;
      int rr = idx >> 2, c = idx & 3;         // 4 chunks per 64B row
      async_load_b128(As + idx * 8,
                      A + (size_t)(rowBase + rr) * K + kt + c * 8);
    }
#pragma unroll
    for (int j = 0; j < 8; ++j) {             // W tile, store transposed (VGPR path)
      int idx = tid * 8 + j;
      int n = idx & 63, k = idx >> 6;
      Ws[n * 32 + k] = W[(size_t)(kt + k) * Nc + colBase + n];
    }
    wait_async0();
    __syncthreads();

    const unsigned short* arow = As + (w * 16 + r) * 32;
    bf16x16 a = frag_a8_8(arow + 16 * h, arow + 16 * h + 8);
#pragma unroll
    for (int cb = 0; cb < 4; ++cb) {
      bf16x16 bfr = frag_contig16(Ws + (cb * 16 + r) * 32 + 16 * h);
      acc[cb] = wmma_bf16(a, bfr, acc[cb]);
    }
    __syncthreads();
  }

#pragma unroll
  for (int cb = 0; cb < 4; ++cb) {
    int n = colBase + cb * 16 + r;
    float bv = bias[n];
#pragma unroll
    for (int rr = 0; rr < 8; ++rr) {
      int m = rowBase + w * 16 + rr + 8 * h;
      Out[(size_t)m * Nc + n] = f32_to_bf16_bits(acc[cb][rr] + bv);
    }
  }
}

// ---------------------------------------------------------------------------
// Kernel 2: energy = q k^T, softmax over keys, write probs fp32 to d_out.
// Block: 64 query rows of one batch. 8 waves = 4 row-groups x 2 key-halves.
// q block + key tiles staged via async DMA (contiguous copies).
// ---------------------------------------------------------------------------
__global__ __launch_bounds__(256) void attn_softmax(
    const unsigned short* __restrict__ Q, const unsigned short* __restrict__ Kmat,
    float* __restrict__ attn) {
  __shared__ unsigned short Qs[64 * 64];    // [qrow][d]
  __shared__ unsigned short Ks[32 * 64];    // [key][d]
  __shared__ float pm[2][64], ps[2][64];
  __shared__ float rowMax[64], rowInv[64];

  const int tid = threadIdx.x, lane = tid & 31, w = tid >> 5;
  const int rg = w >> 1, ch = w & 1;
  const int r = lane & 15, h = lane >> 4;
  const int b = blockIdx.y;
  const int qRow0 = blockIdx.x * 64;

  {  // q block is contiguous in global: flat async copy (8KB = 512 x 16B)
    const unsigned short* qsrc = Q + ((size_t)b * NN + qRow0) * DD;
#pragma unroll
    for (int j = 0; j < 2; ++j) {
      int idx = tid + j * 256;
      async_load_b128(Qs + idx * 8, qsrc + idx * 8);
    }
  }
  wait_async0();
  __syncthreads();

  // q fragments are loop-invariant: build once (D=64 -> two K=32 halves)
  const unsigned short* arow = Qs + (rg * 16 + r) * 64;
  bf16x16 a0 = frag_a8_8(arow + 16 * h, arow + 16 * h + 8);
  bf16x16 a1 = frag_a8_8(arow + 32 + 16 * h, arow + 40 + 16 * h);

  float rm[8], rs[8];
#pragma unroll
  for (int i = 0; i < 8; ++i) { rm[i] = -3.0e38f; rs[i] = 0.0f; }

  const unsigned short* Kb = Kmat + (size_t)b * NN * DD;

  // ---- Pass A: row max & sumexp ----
  for (int kb0 = 0; kb0 < NN; kb0 += 32) {
    async_load_b128(Ks + tid * 8, Kb + (size_t)kb0 * DD + tid * 8);  // 4KB tile
    wait_async0();
    __syncthreads();

    const unsigned short* brow = Ks + (ch * 16 + r) * 64;
    v8f acc = {};
    acc = wmma_bf16(a0, frag_contig16(brow + 16 * h), acc);
    acc = wmma_bf16(a1, frag_contig16(brow + 32 + 16 * h), acc);
#pragma unroll
    for (int rr = 0; rr < 8; ++rr) {
      float e = acc[rr];
      float mn = fmaxf(rm[rr], e);
      rs[rr] = rs[rr] * __expf(rm[rr] - mn) + __expf(e - mn);
      rm[rr] = mn;
    }
    __syncthreads();
  }

  // reduce across the 16 lanes of each half (wave32: xor masks 1..8 stay in-half)
#pragma unroll
  for (int rr = 0; rr < 8; ++rr) {
    float m = rm[rr], s = rs[rr];
#pragma unroll
    for (int off = 1; off < 16; off <<= 1) {
      float mo = __shfl_xor(m, off, 32);
      float so = __shfl_xor(s, off, 32);
      float mn = fmaxf(m, mo);
      s = s * __expf(m - mn) + so * __expf(mo - mn);
      m = mn;
    }
    if (r == 0) {
      int row = rg * 16 + rr + 8 * h;
      pm[ch][row] = m; ps[ch][row] = s;
    }
  }
  __syncthreads();
  if (tid < 64) {  // combine the two key-half waves
    float m0 = pm[0][tid], m1 = pm[1][tid];
    float m = fmaxf(m0, m1);
    float s = ps[0][tid] * __expf(m0 - m) + ps[1][tid] * __expf(m1 - m);
    rowMax[tid] = m;
    rowInv[tid] = 1.0f / s;
  }
  __syncthreads();

  float myMax[8], myInv[8];
#pragma unroll
  for (int rr = 0; rr < 8; ++rr) {
    int row = rg * 16 + rr + 8 * h;
    myMax[rr] = rowMax[row]; myInv[rr] = rowInv[row];
  }

  // ---- Pass B: recompute energy, emit probabilities ----
  for (int kb0 = 0; kb0 < NN; kb0 += 32) {
    async_load_b128(Ks + tid * 8, Kb + (size_t)kb0 * DD + tid * 8);
    wait_async0();
    __syncthreads();

    const unsigned short* brow = Ks + (ch * 16 + r) * 64;
    v8f acc = {};
    acc = wmma_bf16(a0, frag_contig16(brow + 16 * h), acc);
    acc = wmma_bf16(a1, frag_contig16(brow + 32 + 16 * h), acc);

    int col = kb0 + ch * 16 + r;
#pragma unroll
    for (int rr = 0; rr < 8; ++rr) {
      int gRow = qRow0 + rg * 16 + rr + 8 * h;
      attn[((size_t)b * NN + gRow) * (size_t)NN + col] =
          __expf(acc[rr] - myMax[rr]) * myInv[rr];
    }
    __syncthreads();
  }
}

// ---------------------------------------------------------------------------
// Kernel 3: out = gamma * (P @ v) + x.  P fp32 (from d_out), converted to
// bf16 on the LDS-staging load; v bf16 (transposed store). Tile 128x64.
// ---------------------------------------------------------------------------
__global__ __launch_bounds__(256) void pv_residual(
    const float* __restrict__ P, const unsigned short* __restrict__ V,
    const float* __restrict__ x, const float* __restrict__ gamma_p,
    float* __restrict__ out) {
  __shared__ unsigned short Ps[128 * 32];   // [row][k] bf16
  __shared__ unsigned short Vs[64 * 32];    // transposed: [n][k]
  const int tid = threadIdx.x, lane = tid & 31, w = tid >> 5;
  const int r = lane & 15, h = lane >> 4;
  const int b = blockIdx.z;
  const int rowBase = blockIdx.x * 128;
  const int colBase = blockIdx.y * 64;
  v8f acc[4] = {};

  const float* Pb = P + ((size_t)b * NN + rowBase) * (size_t)NN;
  const unsigned short* Vb = V + (size_t)b * NN * CC;

  for (int kt = 0; kt < NN; kt += 32) {
#pragma unroll
    for (int j = 0; j < 16; ++j) {            // 4096 probs, cvt fp32->bf16
      int idx = tid + j * 256;
      int rr = idx >> 5, cc = idx & 31;
      Ps[rr * 32 + cc] = f32_to_bf16_bits(Pb[(size_t)rr * NN + kt + cc]);
    }
#pragma unroll
    for (int j = 0; j < 8; ++j) {             // v tile, store transposed
      int idx = tid * 8 + j;
      int n = idx & 63, k = idx >> 6;
      Vs[n * 32 + k] = Vb[(size_t)(kt + k) * CC + colBase + n];
    }
    __syncthreads();

    const unsigned short* arow = Ps + (w * 16 + r) * 32;
    bf16x16 a = frag_a8_8(arow + 16 * h, arow + 16 * h + 8);
#pragma unroll
    for (int cb = 0; cb < 4; ++cb) {
      bf16x16 bfr = frag_contig16(Vs + (cb * 16 + r) * 32 + 16 * h);
      acc[cb] = wmma_bf16(a, bfr, acc[cb]);
    }
    __syncthreads();
  }

  float g = gamma_p[0];
#pragma unroll
  for (int cb = 0; cb < 4; ++cb) {
    int n = colBase + cb * 16 + r;
#pragma unroll
    for (int rr = 0; rr < 8; ++rr) {
      int m = rowBase + w * 16 + rr + 8 * h;
      size_t o = ((size_t)b * NN + m) * CC + n;
      out[o] = g * acc[cb][rr] + x[o];
    }
  }
}

// ---------------------------------------------------------------------------
extern "C" void kernel_launch(void* const* d_in, const int* in_sizes, int n_in,
                              void* d_out, int out_size, void* d_ws, size_t ws_size,
                              hipStream_t stream) {
  const float* x     = (const float*)d_in[0];
  const float* Wq    = (const float*)d_in[1];
  const float* bq    = (const float*)d_in[2];
  const float* Wk    = (const float*)d_in[3];
  const float* bk    = (const float*)d_in[4];
  const float* Wv    = (const float*)d_in[5];
  const float* bv    = (const float*)d_in[6];
  const float* gamma = (const float*)d_in[7];

  // workspace layout (bf16 buffers), ~145 MB total
  char* ws = (char*)d_ws;
  unsigned short* xb  = (unsigned short*)ws;                         // 64 MB
  unsigned short* wqb = (unsigned short*)(ws + (size_t)64 * 1024 * 1024);
  unsigned short* wkb = wqb + (size_t)CC * DD;                       // 64 KB each
  unsigned short* wvb = wkb + (size_t)CC * DD;
  unsigned short* qb  = wvb + (size_t)CC * CC;                       // 512 KB
  unsigned short* kb  = qb + (size_t)BB * NN * DD;                   // 8 MB each
  unsigned short* vb  = kb + (size_t)BB * NN * DD;                   // then 64 MB

  float* out  = (float*)d_out;                        // [B,N,C]
  float* attn = out + (size_t)BB * NN * CC;           // [B,N,N]

  // 0) fp32 -> bf16 casts
  cast_f32_bf16<<<8192, 256, 0, stream>>>(x,  xb,  (long)BB * NN * CC);
  cast_f32_bf16<<<128,  256, 0, stream>>>(Wq, wqb, (long)CC * DD);
  cast_f32_bf16<<<128,  256, 0, stream>>>(Wk, wkb, (long)CC * DD);
  cast_f32_bf16<<<1024, 256, 0, stream>>>(Wv, wvb, (long)CC * CC);

  // 1) projections: M = B*N = 65536, K = C = 512
  gemm_bias_bf16<<<dim3(65536 / 128, DD / 64), 256, 0, stream>>>(
      xb, wqb, bq, qb, BB * NN, CC, DD);
  gemm_bias_bf16<<<dim3(65536 / 128, DD / 64), 256, 0, stream>>>(
      xb, wkb, bk, kb, BB * NN, CC, DD);
  gemm_bias_bf16<<<dim3(65536 / 128, CC / 64), 256, 0, stream>>>(
      xb, wvb, bv, vb, BB * NN, CC, CC);

  // 2) energy + softmax -> attention probs (fp32, directly into d_out)
  attn_softmax<<<dim3(NN / 64, BB), 256, 0, stream>>>(qb, kb, attn);

  // 3) out = gamma * (P @ v) + x
  pv_residual<<<dim3(NN / 128, CC / 64, BB), 256, 0, stream>>>(
      attn, vb, x, gamma, out);
}
